// RNN_47785806135593
// MI455X (gfx1250) — matrix operand order repaired and checked by reading, not compiled, not run on hardware
//
#include <hip/hip_runtime.h>
#include <stdint.h>

// Problem constants (match reference)
#define Bdim 256
#define Tdim 512
#define Hdim 1024
#define Odim 256

#define NWG      16                 // persistent workgroups; each owns 64 cols of H
#define NTHREADS 256                // 8 wave32 per WG
#define COLS_PER_WG (Hdim / NWG)    // 64

typedef __attribute__((ext_vector_type(16))) __bf16        v16bf;
typedef __attribute__((ext_vector_type(16))) unsigned short v16u;
typedef __attribute__((ext_vector_type(8)))  float          v8f;

union Frag {
    uint4 q[2];   // 32 bytes = 16 bf16 = one WMMA A/B fragment per lane
    v16u  v;
};

__device__ __forceinline__ unsigned short f32_to_bf16(float f) {
    union { float f; unsigned u; } c; c.f = f;
    unsigned u = c.u + 0x7FFFu + ((c.u >> 16) & 1u);   // round-to-nearest-even
    return (unsigned short)(u >> 16);
}

__device__ __forceinline__ float tanh_fast(float v) {
#if defined(__AMDGCN__) && __has_builtin(__builtin_amdgcn_tanhf)
    return __builtin_amdgcn_tanhf(v);
#else
    // tanh(v) = 1 - 2/(exp(2v)+1)   (saturates correctly at +-1)
    float e = __expf(2.0f * v);
    return 1.0f - 2.0f / (e + 1.0f);
#endif
}

// Persistent RNN kernel.
//  - Stage this WG's 1024x64 slice of W_hh (= W_ih rows 1..H) into LDS as
//    pre-swizzled bf16 WMMA B-fragments (128 KB), reused for all 512 steps.
//  - Per step: each wave computes a 32x64 tile of h_new via
//    v_wmma_f32_16x16x32_bf16, adds x_t*w_x + b, applies tanh, stores bf16.
//  - Steps separated by an atomic-counter grid barrier (16 WGs only).
//  - After the last step: out = h_final @ W_ho + b_ho (this WG's 16 cols of O).
__global__ __launch_bounds__(NTHREADS, 1)
void rnn_persistent(const float* __restrict__ x,     // [B,T]
                    const float* __restrict__ W_ih,  // [1+H,H]
                    const float* __restrict__ b_ih,  // [H]
                    const float* __restrict__ W_ho,  // [H,O]
                    const float* __restrict__ b_ho,  // [O]
                    float* __restrict__ out,         // [B,O]
                    unsigned short* __restrict__ h0, // [B,H] bf16
                    unsigned short* __restrict__ h1, // [B,H] bf16
                    unsigned* __restrict__ ctr) {
    // B-fragments for 32 k-tiles x 4 n-tiles x 32 lanes x 32B = 128 KB
    __shared__ uint4 ldsW[32 * 4 * 32 * 2];

    const int tid  = threadIdx.x;
    const int lane = tid & 31;
    const int w    = tid >> 5;          // wave id 0..7
    const int half = lane >> 4;         // 0 or 1
    const int ln   = lane & 15;
    const int g    = blockIdx.x;
    const int col0 = g * COLS_PER_WG;
    const int row0 = w * 32;

    // ---- one-time: W_hh slice -> LDS, swizzled to the B-fragment layout.
    // B element e of lane L holds W_hh[k0 + 16*(L>>4) + e][n0 + (L&15)].
    {
        unsigned short* ldsS = (unsigned short*)ldsW;
        for (int idx = tid; idx < 32 * 4 * 32 * 16; idx += NTHREADS) {
            int e  = idx & 15;
            int la = (idx >> 4) & 31;
            int j  = (idx >> 9) & 3;
            int kt = idx >> 11;
            int k  = kt * 32 + ((la >> 4) << 4) + e;
            int n  = col0 + j * 16 + (la & 15);
            ldsS[idx] = f32_to_bf16(W_ih[(size_t)(1 + k) * Hdim + n]);
        }
    }
    __syncthreads();

    // per-lane input-weight (W_ih row 0) and bias for the 4 n-tiles
    float wx[4], bias[4];
#pragma unroll
    for (int j = 0; j < 4; ++j) {
        int n   = col0 + j * 16 + ln;
        wx[j]   = W_ih[n];
        bias[j] = b_ih[n];
    }

    // LDS base for this lane's B fragments (bumped by 4 KB per k-tile)
    const uint4* const ldsLane = ldsW + (size_t)lane * 2;

    unsigned tgt = 0;
    for (int t = 0; t < Tdim; ++t) {
        const unsigned short* hc = (t & 1) ? h1 : h0;
        unsigned short*       hn = (t & 1) ? h0 : h1;

        // per-step A row pointers (advance 32 bf16 = 4 uint4 per k-tile)
        const uint4* ap0 =
            (const uint4*)(hc + (size_t)(row0 + ln) * Hdim) + 2 * half;
        const uint4* ap1 =
            (const uint4*)(hc + (size_t)(row0 + 16 + ln) * Hdim) + 2 * half;
        const uint4* bp = ldsLane;

        v8f acc[2][4] = {};   // 2 M-tiles x 4 N-tiles of 16x16 f32

#pragma unroll 1              // keep rolled: avoid VGPR spills
        for (int kt = 0; kt < 32; ++kt) {
            Frag a[2];
            a[0].q[0] = ap0[0];
            a[0].q[1] = ap0[1];
            a[1].q[0] = ap1[0];
            a[1].q[1] = ap1[1];
            Frag b[4];
#pragma unroll
            for (int j = 0; j < 4; ++j) {
                b[j].q[0] = bp[j * 64 + 0];
                b[j].q[1] = bp[j * 64 + 1];
            }
#pragma unroll
            for (int mi = 0; mi < 2; ++mi)
#pragma unroll
                for (int j = 0; j < 4; ++j)
                    acc[mi][j] = __builtin_amdgcn_wmma_f32_16x16x32_bf16(
                        false, __builtin_bit_cast(v16bf, a[mi].v),
                        false, __builtin_bit_cast(v16bf, b[j].v),
                        (short)0, acc[mi][j], false, false);
            ap0 += 4;           // next 32 bf16 of this row
            ap1 += 4;
            bp  += 256;         // next k-tile block (4 KB = 256 uint4)
        }

        // epilogue: + x_t * w_x + b, tanh, store bf16 h_new
        float xv[2][8];
#pragma unroll
        for (int mi = 0; mi < 2; ++mi)
#pragma unroll
            for (int r = 0; r < 8; ++r)
                xv[mi][r] = x[(size_t)(row0 + mi * 16 + half * 8 + r) * Tdim + t];

#pragma unroll
        for (int mi = 0; mi < 2; ++mi)
#pragma unroll
            for (int j = 0; j < 4; ++j) {
                const int n = col0 + j * 16 + ln;
#pragma unroll
                for (int r = 0; r < 8; ++r) {
                    float v = acc[mi][j][r] + xv[mi][r] * wx[j] + bias[j];
                    hn[(size_t)(row0 + mi * 16 + half * 8 + r) * Hdim + n] =
                        f32_to_bf16(tanh_fast(v));
                }
            }

        // ---- device-wide step barrier (monotonic counter, 16 WGs)
        __syncthreads();
        tgt += NWG;
        if (tid == 0) {
            __threadfence();                 // release h_new
            atomicAdd(ctr, 1u);
            while (__hip_atomic_load(ctr, __ATOMIC_RELAXED,
                                     __HIP_MEMORY_SCOPE_AGENT) < tgt)
                __builtin_amdgcn_s_sleep(1);
        }
        __syncthreads();
        __threadfence();                     // acquire others' h_new
    }

    // ---- final: out = h_final @ W_ho + b_ho ; this WG owns cols [g*16, g*16+16)
    const unsigned short* hf = h0;           // T even -> final hidden in h0
    const int n0 = g * 16;
    v8f oacc[2] = {};
#pragma unroll 1
    for (int kt = 0; kt < 32; ++kt) {
        const int k0 = kt * 32;
        Frag a[2];
#pragma unroll
        for (int mi = 0; mi < 2; ++mi) {
            const uint4* hp =
                (const uint4*)(hf + (size_t)(row0 + mi * 16 + ln) * Hdim);
            a[mi].q[0] = hp[(k0 >> 3) + 2 * half];
            a[mi].q[1] = hp[(k0 >> 3) + 2 * half + 1];
        }
        Frag bf;
        unsigned short* bs = (unsigned short*)&bf;
#pragma unroll
        for (int e = 0; e < 16; ++e) {
            int k = k0 + half * 16 + e;
            bs[e] = f32_to_bf16(W_ho[(size_t)k * Odim + n0 + ln]);
        }
#pragma unroll
        for (int mi = 0; mi < 2; ++mi)
            oacc[mi] = __builtin_amdgcn_wmma_f32_16x16x32_bf16(
                false, __builtin_bit_cast(v16bf, a[mi].v),
                false, __builtin_bit_cast(v16bf, bf.v),
                (short)0, oacc[mi], false, false);
    }
    const float bo = b_ho[n0 + ln];
#pragma unroll
    for (int mi = 0; mi < 2; ++mi)
#pragma unroll
        for (int r = 0; r < 8; ++r)
            out[(size_t)(row0 + mi * 16 + half * 8 + r) * Odim + n0 + ln] =
                oacc[mi][r] + bo;
}

extern "C" void kernel_launch(void* const* d_in, const int* in_sizes, int n_in,
                              void* d_out, int out_size, void* d_ws, size_t ws_size,
                              hipStream_t stream) {
    (void)in_sizes; (void)n_in; (void)out_size; (void)ws_size;
    const float* x   = (const float*)d_in[0];
    const float* Wih = (const float*)d_in[1];
    const float* bih = (const float*)d_in[2];
    const float* Who = (const float*)d_in[3];
    const float* bho = (const float*)d_in[4];
    float* out = (float*)d_out;

    char* ws = (char*)d_ws;
    const size_t hbytes = (size_t)Bdim * Hdim * 2;   // 512 KB per bf16 h buffer
    unsigned*       ctr = (unsigned*)ws;
    unsigned short* h0  = (unsigned short*)(ws + 256);
    unsigned short* h1  = (unsigned short*)(ws + 256 + hbytes);

    // zero barrier counter + both h buffers (h0 is the t=0 hidden state = 0)
    hipMemsetAsync(d_ws, 0, 256 + 2 * hbytes, stream);

    rnn_persistent<<<dim3(NWG), dim3(NTHREADS), 0, stream>>>(
        x, Wih, bih, Who, bho, out, h0, h1, ctr);
}
